// MDGRU_50087908606505
// MI455X (gfx1250) — matrix operand order repaired and checked by reading, not compiled, not run on hardware
//
#include <hip/hip_runtime.h>
#include <hip/hip_bf16.h>
#include <stdint.h>

// ---------------------------------------------------------------------------
// GRU forward (B=32, T=2048, IN=512, UNITS=512) on gfx1250 (MI455X).
//  prep   : pack kernel/recurrent_kernel into WMMA B-fragment images (bf16),
//           convert X to time-major bf16 [t][b][i].
//  phase 1: xproj = X @ W + bias with v_wmma_f32_16x16x32_bf16, fp32 accum,
//           A/B fragments loaded directly as b128 from pre-packed global.
//  phase 2: persistent 16-WG scan; recurrent weights resident in VGPRs for
//           all 2048 steps; state exchanged as packed bf16 through L2 with
//           atomic spin barriers (2/step: z,r GEMMs -> r*a exchange -> Uh).
// ---------------------------------------------------------------------------

typedef __bf16 v16bf __attribute__((ext_vector_type(16)));
typedef float  v8f   __attribute__((ext_vector_type(8)));

union Frag16 {            // one WMMA 16-bit operand fragment: 16 bf16 = 8 dwords
  v16bf        v;
  unsigned int u[8];
};

__device__ __forceinline__ unsigned short f32_to_bf16(float f) {
  unsigned int u = __float_as_uint(f);
  u += 0x7FFFu + ((u >> 16) & 1u);          // round to nearest even
  return (unsigned short)(u >> 16);
}

__device__ __forceinline__ unsigned int pack_bf16x2(float lo, float hi) {
  return (unsigned int)f32_to_bf16(lo) | ((unsigned int)f32_to_bf16(hi) << 16);
}

__device__ __forceinline__ float sigmoidf_(float x) {
  return 1.0f / (1.0f + __expf(-x));
}

// A-fragment (16x32 bf16, MxK) from a row-major bf16 buffer with row stride
// 512 elements (works for LDS or global pointers). ISA layout: lanes 0-15 row
// M=lane, V0..3 = K pairs {0..7}, V4..7 = {16..23}; lanes 16-31: +8 / +24.
// All 4-dword groups are 16B aligned -> b128 loads.
__device__ __forceinline__ v16bf load_A_frag(const unsigned short* base,
                                             int mbase, int k0, int lane) {
  Frag16 f;
  const int row = mbase + (lane & 15);
  const int khi = (lane >= 16) ? 8 : 0;
  const unsigned short* p = base + row * 512 + k0 + khi;
#pragma unroll
  for (int v = 0; v < 4; ++v) {
    f.u[v]     = *(const unsigned int*)(p + 2 * v);
    f.u[v + 4] = *(const unsigned int*)(p + 16 + 2 * v);
  }
  return f.v;
}

// B-fragment from the pre-packed fragment image: 8 contiguous dwords per lane
// (32B aligned) -> two global_load_b128, zero conversion VALU.
__device__ __forceinline__ v16bf load_B_frag_packed(const unsigned int* pack,
                                                    int frag, int lane) {
  Frag16 f;
  const unsigned int* p = pack + ((size_t)frag * 32 + lane) * 8;
#pragma unroll
  for (int v = 0; v < 8; ++v) f.u[v] = p[v];
  return f.v;
}

__device__ __forceinline__ v8f wmma_bf16(v16bf a, v16bf b, v8f c) {
  return __builtin_amdgcn_wmma_f32_16x16x32_bf16(false, a, false, b,
                                                 (short)0, c, false, false);
}

// Monotonic device-scope barrier across the 16 persistent workgroups.
__device__ __forceinline__ void grid_barrier(int* cnt, int target) {
  __syncthreads();
  if (threadIdx.x == 0) {
    __threadfence();
    atomicAdd(cnt, 1);
    while (__hip_atomic_load(cnt, __ATOMIC_ACQUIRE, __HIP_MEMORY_SCOPE_AGENT) < target) {
      __builtin_amdgcn_s_sleep(1);
    }
  }
  __syncthreads();
  __threadfence();
}

// ---------------------------------------------------------------------------
// Prep: pack a 512x1536 fp32 matrix into the WMMA B-fragment register image.
// frag f = colt*16 + kf (colt 0..95, kf 0..15); dword (f, lane, v) =
// pack(bf16(M[kg+2v, col]), bf16(M[kg+2v+1, col])), col = colt*16 + (lane&15),
// kg = kf*32 + (lane>=16 ? 8 : 0) + (v>=4 ? 8 : 0).  One thread per output dword.
// ---------------------------------------------------------------------------
__global__ void pack_weights_kernel(const float* __restrict__ W,
                                    const float* __restrict__ RK,
                                    unsigned int* __restrict__ Wpack,
                                    unsigned int* __restrict__ RKpack) {
  const int idx = blockIdx.x * 256 + threadIdx.x;      // 0..393215
  const float* src = blockIdx.y ? RK : W;
  unsigned int* dst = blockIdx.y ? RKpack : Wpack;
  const int v = idx & 7, lane = (idx >> 3) & 31, f = idx >> 8;
  const int col = (f >> 4) * 16 + (lane & 15);
  const int k = (f & 15) * 32 + ((lane >= 16) ? 8 : 0) + 2 * v + ((v >= 4) ? 8 : 0);
  dst[idx] = pack_bf16x2(src[(size_t)k * 1536 + col],
                         src[(size_t)(k + 1) * 1536 + col]);
}

// X[b][t][i] fp32 -> xbf[t][b][i] bf16 (packed pairs, one dword per thread).
__global__ void convert_X_kernel(const float* __restrict__ X,
                                 unsigned int* __restrict__ xbf) {
  const int idx = blockIdx.x * 256 + threadIdx.x;      // 0..16777215
  const int t = idx >> 13;
  const int b = (idx >> 8) & 31;
  const int ip = idx & 255;
  const float* p = X + ((size_t)b * 2048 + t) * 512 + 2 * ip;
  xbf[idx] = pack_bf16x2(p[0], p[1]);
}

// Zero initial state a0 (packed bf16 slot 0) and the barrier counter.
__global__ void init_kernel(unsigned int* aBuf, int* cnt) {
  const int i = blockIdx.x * blockDim.x + threadIdx.x;
  if (i == 0) *cnt = 0;
  if (i < 8192) aBuf[i] = 0u;
}

// ---------------------------------------------------------------------------
// Phase 1: xproj[t][b][g] = X[b][t][:] @ W[:,g] + bias[g]
// grid = (24 ntiles of 64 cols, 2048 t), block = 256 (8 waves), no LDS.
// Each wave: one 16x16 tile, K=512 -> 16 WMMAs; A from xbf, B from Wpack.
// ---------------------------------------------------------------------------
__global__ __launch_bounds__(256) void xproj_kernel(const unsigned short* __restrict__ xbf,
                                                    const unsigned int* __restrict__ Wpack,
                                                    const float* __restrict__ bias,
                                                    float* __restrict__ xproj) {
  const int t = blockIdx.y, ntile = blockIdx.x;
  const int wave = threadIdx.x >> 5, lane = threadIdx.x & 31;
  const int mt = wave >> 2;                 // batch tile 0..1
  const int nt = wave & 3;                  // 16-col tile 0..3
  const int colt = ntile * 4 + nt;          // global 16-col tile 0..95
  const unsigned short* Abase = xbf + (size_t)t * 32 * 512;

  v8f c = {};
#pragma unroll
  for (int kf = 0; kf < 16; ++kf) {
    v16bf a = load_A_frag(Abase, mt * 16, kf * 32, lane);
    v16bf b = load_B_frag_packed(Wpack, colt * 16 + kf, lane);
    c = wmma_bf16(a, b, c);
  }

  const int mycol = colt * 16 + (lane & 15);
  const float bb = bias[mycol];
  const int mo = (lane >= 16) ? 8 : 0;
#pragma unroll
  for (int r = 0; r < 8; ++r) {
    const int bidx = mt * 16 + r + mo;
    xproj[((size_t)t * 32 + bidx) * 1536 + mycol] = c[r] + bb;
  }
}

// ---------------------------------------------------------------------------
// Phase 2: persistent scan. grid = 16 WGs, block = 384 (12 waves).
// WG w owns UNITS columns [32w, 32w+32). gate = wave/4 (0=z,1=r,2=h),
// mt = (wave>>1)&1, nt = wave&1. State exchanged as packed bf16 dwords;
// each WG's own fp32 state slice lives persistently in LDS.
// ---------------------------------------------------------------------------
__global__ __launch_bounds__(384) void gru_scan_kernel(const unsigned int* __restrict__ RKpack,
                                                       const float* __restrict__ xproj,
                                                       unsigned int* __restrict__ aBuf,
                                                       unsigned int* __restrict__ raBuf,
                                                       int* __restrict__ cnt,
                                                       float* __restrict__ out) {
  __shared__ __align__(16) unsigned int sBF[32 * 256];  // 32 KB: bf16 [32][512]
  __shared__ float sOwnA[32 * 32];                      // own fp32 state slice
  __shared__ float sTile[12 * 256];                     // per-wave gate tiles

  const int wg = blockIdx.x;
  const int c0 = wg * 32;
  const int tid = threadIdx.x;
  const int wave = tid >> 5, lane = tid & 31;
  const int gate = wave >> 2;
  const int mt = (wave >> 1) & 1;
  const int nt = wave & 1;

  // Recurrent weights resident in VGPRs for the whole scan (128 VGPRs/wave).
  Frag16 Bf[16];
  const int colt = gate * 32 + wg * 2 + nt;             // 16-col tile in RK
#pragma unroll
  for (int kf = 0; kf < 16; ++kf)
    Bf[kf].v = load_B_frag_packed(RKpack, colt * 16 + kf, lane);

  for (int i = tid; i < 1024; i += 384) sOwnA[i] = 0.0f;   // a0 = 0

  unsigned int* aCur = aBuf;
  unsigned int* aNxt = aBuf + 8192;

  for (int t = 0; t < 2048; ++t) {
    // Stage full state a (packed bf16) into LDS: straight dword copy.
    for (int d = tid; d < 8192; d += 384) sBF[d] = aCur[d];
    __syncthreads();

    // Pz, Pr = a @ {Uz,Ur}[:, own cols]  (waves 0..7)
    if (wave < 8) {
      v8f c = {};
#pragma unroll
      for (int kf = 0; kf < 16; ++kf) {
        v16bf a = load_A_frag((const unsigned short*)sBF, mt * 16, kf * 32, lane);
        c = wmma_bf16(a, Bf[kf].v, c);
      }
      const int n = lane & 15, mo = (lane >= 16) ? 8 : 0;
#pragma unroll
      for (int r = 0; r < 8; ++r) sTile[wave * 256 + (r + mo) * 16 + n] = c[r];
    }
    __syncthreads();

    // z = sigmoid(Pz+xz), r = sigmoid(Pr+xr); publish r*a (bf16); stash z.
    for (int idx = tid; idx < 512; idx += 384) {
      const int b = idx >> 4, j = (idx & 15) * 2;       // column pair j, j+1
      const int mti = b >> 4, nti = j >> 4;
      const int m = b & 15, n = j & 15;
      const int wz = mti * 2 + nti, wr = 4 + mti * 2 + nti;
      const size_t xb = ((size_t)t * 32 + b) * 1536 + (size_t)(c0 + j);
      const float z0 = sigmoidf_(sTile[wz * 256 + m * 16 + n]     + xproj[xb]);
      const float z1 = sigmoidf_(sTile[wz * 256 + m * 16 + n + 1] + xproj[xb + 1]);
      const float r0 = sigmoidf_(sTile[wr * 256 + m * 16 + n]     + xproj[xb + 512]);
      const float r1 = sigmoidf_(sTile[wr * 256 + m * 16 + n + 1] + xproj[xb + 513]);
      sTile[wz * 256 + m * 16 + n]     = z0;
      sTile[wz * 256 + m * 16 + n + 1] = z1;
      raBuf[b * 256 + ((c0 + j) >> 1)] =
          pack_bf16x2(r0 * sOwnA[b * 32 + j], r1 * sOwnA[b * 32 + j + 1]);
    }
    __threadfence();
    grid_barrier(cnt, 16 * (2 * t + 1));

    // Stage full (r*a) into LDS.
    for (int d = tid; d < 8192; d += 384) sBF[d] = raBuf[d];
    __syncthreads();

    // Ph = (r*a) @ Uh[:, own cols]  (waves 8..11)
    if (wave >= 8) {
      v8f c = {};
#pragma unroll
      for (int kf = 0; kf < 16; ++kf) {
        v16bf a = load_A_frag((const unsigned short*)sBF, mt * 16, kf * 32, lane);
        c = wmma_bf16(a, Bf[kf].v, c);
      }
      const int n = lane & 15, mo = (lane >= 16) ? 8 : 0;
#pragma unroll
      for (int r = 0; r < 8; ++r) sTile[wave * 256 + (r + mo) * 16 + n] = c[r];
    }
    __syncthreads();

    // h = tanh(Ph + xh); a_next = z*a + (1-z)*h (own columns only).
    for (int idx = tid; idx < 512; idx += 384) {
      const int b = idx >> 4, j = (idx & 15) * 2;
      const int mti = b >> 4, nti = j >> 4;
      const int m = b & 15, n = j & 15;
      const int wz = mti * 2 + nti, wh = 8 + mti * 2 + nti;
      const size_t xb = ((size_t)t * 32 + b) * 1536 + (size_t)(c0 + j);
      const float h0 = tanhf(sTile[wh * 256 + m * 16 + n]     + xproj[xb + 1024]);
      const float h1 = tanhf(sTile[wh * 256 + m * 16 + n + 1] + xproj[xb + 1025]);
      const float z0 = sTile[wz * 256 + m * 16 + n];
      const float z1 = sTile[wz * 256 + m * 16 + n + 1];
      const float a0 = z0 * sOwnA[b * 32 + j]     + (1.0f - z0) * h0;
      const float a1 = z1 * sOwnA[b * 32 + j + 1] + (1.0f - z1) * h1;
      sOwnA[b * 32 + j]     = a0;
      sOwnA[b * 32 + j + 1] = a1;
      aNxt[b * 256 + ((c0 + j) >> 1)] = pack_bf16x2(a0, a1);
    }
    __threadfence();
    grid_barrier(cnt, 16 * (2 * t + 2));

    unsigned int* tmp = aCur; aCur = aNxt; aNxt = tmp;
  }

  // Final state: own fp32 slice is already in LDS.
  __syncthreads();
  for (int idx = tid; idx < 1024; idx += 384) {
    const int b = idx >> 5, j = idx & 31;
    out[b * 512 + c0 + j] = sOwnA[b * 32 + j];
  }
}

// ---------------------------------------------------------------------------
// Host launch. Inputs (setup_inputs order): X, kernel, recurrent_kernel, bias.
// Workspace layout (dwords after the fp32 xproj block):
//   xproj  : 2048*32*1536 fp32                  (384 MB)
//   xbf    : 16,777,216 dw  (X as bf16, t-major) (64 MB)
//   Wpack  : 393,216 dw  (B-fragment image)     (1.5 MB)
//   RKpack : 393,216 dw                          (1.5 MB)
//   aBuf   : 2 x 8192 dw (state ping-pong, bf16)
//   raBuf  : 8192 dw (r*a exchange, bf16)
//   cnt    : 1 dw (barrier counter)
// ---------------------------------------------------------------------------
extern "C" void kernel_launch(void* const* d_in, const int* in_sizes, int n_in,
                              void* d_out, int out_size, void* d_ws, size_t ws_size,
                              hipStream_t stream) {
  const float* X    = (const float*)d_in[0];
  const float* W    = (const float*)d_in[1];
  const float* RK   = (const float*)d_in[2];
  const float* bias = (const float*)d_in[3];
  float* out = (float*)d_out;

  float* xproj = (float*)d_ws;
  const size_t xcount = (size_t)2048 * 32 * 1536;
  unsigned int* xbf    = (unsigned int*)(xproj + xcount);
  unsigned int* Wpack  = xbf + (size_t)16777216;
  unsigned int* RKpack = Wpack + 393216;
  unsigned int* aBuf   = RKpack + 393216;
  unsigned int* raBuf  = aBuf + 2 * 8192;
  int*          cnt    = (int*)(raBuf + 8192);

  init_kernel<<<dim3(64), dim3(256), 0, stream>>>(aBuf, cnt);
  pack_weights_kernel<<<dim3(1536, 2), dim3(256), 0, stream>>>(W, RK, Wpack, RKpack);
  convert_X_kernel<<<dim3(65536), dim3(256), 0, stream>>>(X, xbf);
  xproj_kernel<<<dim3(24, 2048), dim3(256), 0, stream>>>(
      (const unsigned short*)xbf, Wpack, bias, xproj);
  gru_scan_kernel<<<dim3(16), dim3(384), 0, stream>>>(
      RKpack, xproj, aBuf, raBuf, cnt, out);
}
